// AttnBlock_72138270704219
// MI455X (gfx1250) — matrix operand order repaired and checked by reading, not compiled, not run on hardware
//
#include <hip/hip_runtime.h>
#include <stdint.h>

#define BATCH   8
#define CH      512
#define HW      4096
#define NGROUPS 32
#define CPG     16
#define SCALE_ATTN 0.044194173824159216f  // 1/sqrt(512)

typedef __attribute__((ext_vector_type(16))) __bf16       v16bf;
typedef __attribute__((ext_vector_type(8)))  float        v8f;
typedef __attribute__((ext_vector_type(4)))  unsigned int u32x4;

union Frag {
  u32x4          q[2];
  unsigned int   u[8];
  unsigned short h[16];
  v16bf          v;
};
union Acc8 {
  v8f   v;
  float f[8];
};

// native bf16 convert (single v_cvt on gfx1250) instead of integer emulation
__device__ __forceinline__ unsigned short f2bf(float f) {
  return __builtin_bit_cast(unsigned short, (__bf16)f);
}

__device__ __forceinline__ v8f wmma_bf16(const Frag& a, const Frag& b, v8f c) {
  return __builtin_amdgcn_wmma_f32_16x16x32_bf16(false, a.v, false, b.v,
                                                 (short)0, c, false, false);
}

// CDNA5 async memory->LDS DMA (tracked by ASYNCcnt). The LDS destination is
// the low-32 bits of the generic pointer (LDS aperture: addr[31:0] == offset).
__device__ __forceinline__ void async_load_b128(const void* gptr, void* lptr) {
  unsigned loff = (unsigned)(size_t)lptr;
  asm volatile("global_load_async_to_lds_b128 %0, %1, off"
               :: "v"(loff), "v"(gptr) : "memory");
}
__device__ __forceinline__ void wait_async0() {
  asm volatile("s_wait_asynccnt 0x0" ::: "memory");
}

__device__ __forceinline__ float half16_max(float v) {
#pragma unroll
  for (int off = 8; off >= 1; off >>= 1) v = fmaxf(v, __shfl_xor(v, off, 32));
  return v;
}
__device__ __forceinline__ float half16_sum(float v) {
#pragma unroll
  for (int off = 8; off >= 1; off >>= 1) v += __shfl_xor(v, off, 32);
  return v;
}

// ---------------------------------------------------------------------------
// Kernel 1: GroupNorm (B,C,H,W) -> token-major bf16 tokens [B][HW][C]
// Pass 2 emits packed 8-channel b128 stores per token.
// ---------------------------------------------------------------------------
__global__ __launch_bounds__(256) void gnorm_tok(
    const float* __restrict__ x,
    const float* __restrict__ gw,
    const float* __restrict__ gb,
    unsigned short* __restrict__ t)
{
  const int g = blockIdx.x, b = blockIdx.y;
  const int tid = threadIdx.x;
  const float* xg = x + ((size_t)b * CH + g * CPG) * HW;

  float s = 0.f, s2 = 0.f;
  for (int i = tid; i < CPG * HW; i += 256) {
    float v = xg[i];
    s += v; s2 += v * v;
  }
#pragma unroll
  for (int off = 16; off >= 1; off >>= 1) {
    s  += __shfl_xor(s,  off, 32);
    s2 += __shfl_xor(s2, off, 32);
  }
  __shared__ float rs[8], rs2[8], stat[2];
  if ((tid & 31) == 0) { rs[tid >> 5] = s; rs2[tid >> 5] = s2; }
  __syncthreads();
  if (tid == 0) {
    float ts = 0.f, ts2 = 0.f;
    for (int i = 0; i < 8; i++) { ts += rs[i]; ts2 += rs2[i]; }
    const float inv_n = 1.f / (float)(CPG * HW);
    float mean = ts * inv_n;
    float var  = ts2 * inv_n - mean * mean;
    stat[0] = mean;
    stat[1] = rsqrtf(var + 1e-5f);
  }
  __syncthreads();
  const float mean = stat[0], rstd = stat[1];

  // each thread owns a fixed 8-channel half-group -> packed stores
  const int c0 = (tid & 1) * 8;
  float sc8[8], of8[8];
#pragma unroll
  for (int j = 0; j < 8; j++) {
    const int cg = g * CPG + c0 + j;
    const float a = gw[cg] * rstd;
    sc8[j] = a;
    of8[j] = gb[cg] - mean * a;
  }
  unsigned short* tb = t + (size_t)b * HW * CH + g * CPG + c0;
  for (int it = 0; it < 32; it++) {
    const int hw = (tid >> 1) + it * 128;
    Frag pk;
#pragma unroll
    for (int j = 0; j < 8; j++)
      pk.h[j] = f2bf(xg[(size_t)(c0 + j) * HW + hw] * sc8[j] + of8[j]);
    *(u32x4*)(tb + (size_t)hw * CH) = pk.q[0];
  }
}

// ---------------------------------------------------------------------------
// Kernel 2: token GEMM  out[m][n] = sum_k A_bf16[m][k] * W_f32[n][k] + bias[n]
// Double-buffered LDS tiles; A staged via async-to-LDS DMA, W prefetched to
// registers during WMMA phase and converted to bf16 afterwards.
// EPI==0: bf16 token-major out (QKV).  EPI==1: f32 NCHW out + residual.
// ---------------------------------------------------------------------------
template<int EPI>
__global__ __launch_bounds__(256) void gemm_tok(
    const unsigned short* __restrict__ A,
    const float* __restrict__ W,
    const float* __restrict__ bias,
    const float* __restrict__ resid,
    unsigned short* __restrict__ outb,
    float* __restrict__ outf)
{
  __shared__ __align__(16) unsigned short As[2][128 * 32];   // 16 KB
  __shared__ __align__(16) unsigned short Ws[2][128 * 32];   // 16 KB

  const int tid  = threadIdx.x;
  const int b    = blockIdx.z;
  const int m0   = blockIdx.x * 128;
  const int n0   = blockIdx.y * 128;
  const int w    = tid >> 5, lane = tid & 31;
  const int wm   = (w & 3) * 32, wn = (w >> 2) * 64;
  const int lm   = lane & 15,   lh = lane >> 4;

  const int ar0 = tid >> 1, ac0 = (tid & 1) * 16;
  const size_t abase = ((size_t)b * HW + m0) * CH;
  const float* wrow = W + (size_t)(n0 + ar0) * CH + ac0;

  Acc8 acc[2][4];
#pragma unroll
  for (int i = 0; i < 2; i++)
#pragma unroll
    for (int j = 0; j < 4; j++)
#pragma unroll
      for (int r = 0; r < 8; r++) acc[i][j].f[r] = 0.f;

  // ---- prologue: stage tile 0 ----
#pragma unroll
  for (int j = 0; j < 2; j++) {
    int cc = tid + j * 256;
    int r = cc >> 2, c0 = (cc & 3) * 8;
    async_load_b128(A + abase + (size_t)r * CH + c0, &As[0][r * 32 + c0]);
  }
  {
    Frag t0;
#pragma unroll
    for (int i = 0; i < 16; i++) t0.h[i] = f2bf(wrow[i]);
    *(u32x4*)(&Ws[0][ar0 * 32 + ac0])     = t0.q[0];
    *(u32x4*)(&Ws[0][ar0 * 32 + ac0 + 8]) = t0.q[1];
  }
  wait_async0();
  __syncthreads();

  int p = 0;
  for (int kb = 0; kb < CH; kb += 32, p ^= 1) {
    const bool pre = (kb + 32) < CH;
    float wreg[16];
    if (pre) {
#pragma unroll
      for (int j = 0; j < 2; j++) {
        int cc = tid + j * 256;
        int r = cc >> 2, c0 = (cc & 3) * 8;
        async_load_b128(A + abase + (size_t)r * CH + kb + 32 + c0,
                        &As[p ^ 1][r * 32 + c0]);
      }
#pragma unroll
      for (int i = 0; i < 16; i++) wreg[i] = wrow[kb + 32 + i];
    }

    // ---- WMMA phase on buffer p ----
    Frag af[2];
#pragma unroll
    for (int ms = 0; ms < 2; ms++) {
      const unsigned short* pa = &As[p][(wm + ms * 16 + lm) * 32 + lh * 8];
      af[ms].q[0] = *(const u32x4*)pa;
      af[ms].q[1] = *(const u32x4*)(pa + 16);
    }
#pragma unroll
    for (int ns = 0; ns < 4; ns++) {
      Frag bfv;
      const unsigned short* pb = &Ws[p][(wn + ns * 16 + lm) * 32 + lh * 8];
      bfv.q[0] = *(const u32x4*)pb;
      bfv.q[1] = *(const u32x4*)(pb + 16);
#pragma unroll
      for (int ms = 0; ms < 2; ms++)
        acc[ms][ns].v = wmma_bf16(af[ms], bfv, acc[ms][ns].v);
    }

    if (pre) {
      Frag tw;
#pragma unroll
      for (int i = 0; i < 16; i++) tw.h[i] = f2bf(wreg[i]);
      *(u32x4*)(&Ws[p ^ 1][ar0 * 32 + ac0])     = tw.q[0];
      *(u32x4*)(&Ws[p ^ 1][ar0 * 32 + ac0 + 8]) = tw.q[1];
      wait_async0();
    }
    __syncthreads();
  }

  // ---- epilogue ----
#pragma unroll
  for (int ms = 0; ms < 2; ms++) {
#pragma unroll
    for (int ns = 0; ns < 4; ns++) {
      const int mg = m0 + wm + ms * 16 + lh * 8;
      const int ng = n0 + wn + ns * 16 + lm;
      const float bn = bias[ng];
#pragma unroll
      for (int r = 0; r < 8; r++) {
        float v = acc[ms][ns].f[r] + bn;
        if (EPI == 0) {
          outb[((size_t)b * HW + mg + r) * CH + ng] = f2bf(v);
        } else {
          size_t idx = ((size_t)b * CH + ng) * HW + mg + r;
          outf[idx] = resid[idx] + v;
        }
      }
    }
  }
}

// ---------------------------------------------------------------------------
// Kernel 3: flash attention, 8 waves x 16 queries = 128 queries / WG.
// K tile double-buffered + V tile prefetched one block ahead, both staged with
// async-to-LDS DMA overlapping the WMMA stream. V transposed LDS->LDS so both
// GEMMs' B-fragments are contiguous ds_load_b128s.
// ---------------------------------------------------------------------------
__global__ __launch_bounds__(256) void attn_flash(
    const unsigned short* __restrict__ Q,
    const unsigned short* __restrict__ K,
    const unsigned short* __restrict__ V,
    unsigned short* __restrict__ O)
{
  __shared__ __align__(16) unsigned short Kbuf[2][32 * CH];  // 64 KB
  __shared__ __align__(16) unsigned short Vraw[32 * CH];     // 32 KB
  __shared__ __align__(16) unsigned short VT[32 * CH];       // 32 KB
  __shared__ __align__(16) unsigned short Psh[8 * 512];      // 8 KB

  const int tid  = threadIdx.x;
  const int b    = blockIdx.y;
  const int w    = tid >> 5, lane = tid & 31;
  const int lm   = lane & 15, lh = lane >> 4;
  const int q0   = blockIdx.x * 128 + w * 16;
  const size_t kvbase = (size_t)b * HW * CH;

  // Q fragments: the wave's 16 queries, all 512 channels (16 chunks of 32)
  Frag qf[16];
  {
    const unsigned short* qp = Q + ((size_t)b * HW + q0 + lm) * CH + lh * 8;
#pragma unroll
    for (int kc = 0; kc < 16; kc++) {
      qf[kc].q[0] = *(const u32x4*)(qp + kc * 32);
      qf[kc].q[1] = *(const u32x4*)(qp + kc * 32 + 16);
    }
  }

  Acc8 oacc[32];
#pragma unroll
  for (int nt = 0; nt < 32; nt++)
#pragma unroll
    for (int r = 0; r < 8; r++) oacc[nt].f[r] = 0.f;

  float mrow[8], lrow[8];
#pragma unroll
  for (int r = 0; r < 8; r++) { mrow[r] = -1e30f; lrow[r] = 0.f; }

  unsigned short* Pw = Psh + w * 512;

  // ---- prologue: async-stage K[0] and V[0] ----
#pragma unroll
  for (int j = 0; j < 8; j++) {
    int cc = tid + j * 256;
    int key = cc >> 6, c0 = (cc & 63) * 8;
    async_load_b128(K + kvbase + (size_t)key * CH + c0, &Kbuf[0][key * CH + c0]);
    async_load_b128(V + kvbase + (size_t)key * CH + c0, &Vraw[key * CH + c0]);
  }

  int blk = 0;
  for (int kb = 0; kb < HW; kb += 32, blk ^= 1) {
    wait_async0();        // K[i] and Vraw[i] resident
    __syncthreads();      // + all waves past last block's reads
    const unsigned short* Kc = Kbuf[blk];
    const bool pre = (kb + 32) < HW;

    // issue next K tile DMA into the other buffer
    if (pre) {
#pragma unroll
      for (int j = 0; j < 8; j++) {
        int cc = tid + j * 256;
        int key = cc >> 6, c0 = (cc & 63) * 8;
        async_load_b128(K + kvbase + (size_t)(kb + 32 + key) * CH + c0,
                        &Kbuf[blk ^ 1][key * CH + c0]);
      }
    }
    // transpose Vraw[i] -> VT (pure LDS traffic)
#pragma unroll
    for (int j = 0; j < 8; j++) {
      int cc = tid + j * 256;
      int key = cc >> 6, c0 = (cc & 63) * 8;
      Frag tmp;
      tmp.q[0] = *(const u32x4*)(Vraw + key * CH + c0);
#pragma unroll
      for (int jj = 0; jj < 8; jj++)
        VT[(c0 + jj) * 32 + key] = tmp.h[jj];
    }
    __syncthreads();      // VT ready, Vraw free for the next DMA
    if (pre) {
#pragma unroll
      for (int j = 0; j < 8; j++) {
        int cc = tid + j * 256;
        int key = cc >> 6, c0 = (cc & 63) * 8;
        async_load_b128(V + kvbase + (size_t)(kb + 32 + key) * CH + c0,
                        &Vraw[key * CH + c0]);
      }
    }

    // ---- S = Q @ K^T  (16 queries x 32 keys) ----
    Acc8 s0, s1;
#pragma unroll
    for (int r = 0; r < 8; r++) { s0.f[r] = 0.f; s1.f[r] = 0.f; }
#pragma unroll
    for (int kc = 0; kc < 16; kc++) {
      Frag b0, b1;
      const unsigned short* p0 = Kc + lm * CH + kc * 32 + lh * 8;
      b0.q[0] = *(const u32x4*)p0;
      b0.q[1] = *(const u32x4*)(p0 + 16);
      const unsigned short* p1 = p0 + 16 * CH;
      b1.q[0] = *(const u32x4*)p1;
      b1.q[1] = *(const u32x4*)(p1 + 16);
      s0.v = wmma_bf16(qf[kc], b0, s0.v);
      s1.v = wmma_bf16(qf[kc], b1, s1.v);
    }

    // ---- online softmax ----
    float corr[8];
#pragma unroll
    for (int r = 0; r < 8; r++) {
      float mx = half16_max(fmaxf(s0.f[r], s1.f[r]));
      float mn = fmaxf(mrow[r], mx);
      corr[r] = __expf((mrow[r] - mn) * SCALE_ATTN);
      mrow[r] = mn;
      float p0 = __expf((s0.f[r] - mn) * SCALE_ATTN);
      float p1 = __expf((s1.f[r] - mn) * SCALE_ATTN);
      s0.f[r] = p0; s1.f[r] = p1;
      lrow[r] = lrow[r] * corr[r] + half16_sum(p0 + p1);
    }
    // P tile: C-layout -> [m][key] so reload matches A-layout (same-wave LDS)
#pragma unroll
    for (int r = 0; r < 8; r++) {
      Pw[(r + lh * 8) * 32 + lm]      = f2bf(s0.f[r]);
      Pw[(r + lh * 8) * 32 + 16 + lm] = f2bf(s1.f[r]);
    }
#pragma unroll
    for (int nt = 0; nt < 32; nt++)
#pragma unroll
      for (int r = 0; r < 8; r++) oacc[nt].f[r] *= corr[r];

    // ---- O += P @ V ----
    Frag pf;
    {
      const unsigned short* pp = Pw + lm * 32 + lh * 8;
      pf.q[0] = *(const u32x4*)pp;
      pf.q[1] = *(const u32x4*)(pp + 16);
    }
#pragma unroll
    for (int nt = 0; nt < 32; nt++) {
      Frag bv;
      const unsigned short* pv = VT + (nt * 16 + lm) * 32 + lh * 8;
      bv.q[0] = *(const u32x4*)pv;
      bv.q[1] = *(const u32x4*)(pv + 16);
      oacc[nt].v = wmma_bf16(pf, bv, oacc[nt].v);
    }
  }

  // ---- finalize: O /= l, store bf16 token-major ----
  float inv[8];
#pragma unroll
  for (int r = 0; r < 8; r++) inv[r] = 1.f / lrow[r];
  unsigned short* op = O + ((size_t)b * HW + q0) * CH;
#pragma unroll
  for (int nt = 0; nt < 32; nt++)
#pragma unroll
    for (int r = 0; r < 8; r++)
      op[(size_t)(r + lh * 8) * CH + nt * 16 + lm] =
          f2bf(oacc[nt].f[r] * inv[r]);
}

// ---------------------------------------------------------------------------
extern "C" void kernel_launch(void* const* d_in, const int* in_sizes, int n_in,
                              void* d_out, int out_size, void* d_ws, size_t ws_size,
                              hipStream_t stream)
{
  (void)in_sizes; (void)n_in; (void)out_size; (void)ws_size;
  const float* x   = (const float*)d_in[0];
  const float* gnw = (const float*)d_in[1];
  const float* gnb = (const float*)d_in[2];
  const float* wq  = (const float*)d_in[3];
  const float* bq  = (const float*)d_in[4];
  const float* wk  = (const float*)d_in[5];
  const float* bk  = (const float*)d_in[6];
  const float* wv  = (const float*)d_in[7];
  const float* bv  = (const float*)d_in[8];
  const float* wo  = (const float*)d_in[9];
  const float* bo  = (const float*)d_in[10];
  float* out = (float*)d_out;

  const size_t nelem = (size_t)BATCH * HW * CH;
  unsigned short* tbuf = (unsigned short*)d_ws;
  unsigned short* qbuf = tbuf + nelem;
  unsigned short* kbuf = qbuf + nelem;
  unsigned short* vbuf = kbuf + nelem;
  unsigned short* obuf = tbuf;   // t is dead after QKV; reuse for attention out

  gnorm_tok<<<dim3(NGROUPS, BATCH), 256, 0, stream>>>(x, gnw, gnb, tbuf);

  dim3 ggrid(HW / 128, CH / 128, BATCH);
  gemm_tok<0><<<ggrid, 256, 0, stream>>>(tbuf, wq, bq, nullptr, qbuf, nullptr);
  gemm_tok<0><<<ggrid, 256, 0, stream>>>(tbuf, wk, bk, nullptr, kbuf, nullptr);
  gemm_tok<0><<<ggrid, 256, 0, stream>>>(tbuf, wv, bv, nullptr, vbuf, nullptr);

  attn_flash<<<dim3(HW / 128, BATCH), 256, 0, stream>>>(qbuf, kbuf, vbuf, obuf);

  gemm_tok<1><<<ggrid, 256, 0, stream>>>(obuf, wo, bo, x, nullptr, out);
}
